// UResNet_2894807958206
// MI455X (gfx1250) — matrix operand order repaired and checked by reading, not compiled, not run on hardware
//
#include <hip/hip_runtime.h>

typedef __attribute__((ext_vector_type(16))) _Float16 v16h;
typedef __attribute__((ext_vector_type(8)))  float    v8f;

#define NSTR 5
static const int NP[NSTR] = {16, 32, 48, 64, 80};

// ---------------------------------------------------------------------------
// Gather -> WMMA(f16 in, f32 acc) -> scatter-add.
//   * One wave = 16 rulebook pairs x 16 output channels; K chunked by 32.
//   * Weight (B) fragments are converted to f16 and pre-swizzled into LDS once
//     per block (shared by all 8 waves) -> two ds_load_b128 per chunk.
//   * A gathers use float4 (global_load_b128) on full chunks; masked scalar
//     path only for ragged K tails / out-of-range rows.
//   * Output indices are unique within one launch (rulebook property), so the
//     += scatter is conflict-free and deterministic (no atomics).
// ---------------------------------------------------------------------------
__global__ void __launch_bounds__(256) k_spconv_wmma(
    const float* __restrict__ X, const float* __restrict__ W,
    const int* __restrict__ gi, const int* __restrict__ so,
    float* __restrict__ OUT, int npairs, int Cin, int Cout)
{
  __shared__ alignas(32) _Float16 sB[4 * 512];   // up to 4 K-chunks (Cin<=128)

  const int lane  = threadIdx.x & 31;            // wave32
  const int wave  = threadIdx.x >> 5;
  const int half  = lane >> 4;
  const int l15   = lane & 15;
  const int cout0 = blockIdx.y * 16;
  const int chunks = (Cin + 31) >> 5;

  // ---- cooperative B staging: fragment element (lane,j) <-> (K,N) ---------
  const int totB = chunks << 9;                  // chunks * 512 f16
  for (int e = threadIdx.x; e < totB; e += 256) {
    int chunk  = e >> 9;
    int within = e & 511;
    int blane  = within >> 4;
    int j      = within & 15;
    int c      = (chunk << 5) + ((j >> 3) << 4) + ((blane >> 4) << 3) + (j & 7);   // K index
    int n      = blane & 15;                               // N index
    float v = (c < Cin) ? W[(long)c * Cout + cout0 + n] : 0.f;
    sB[e] = (_Float16)v;
  }
  __syncthreads();

  const int tile = blockIdx.x * 8 + wave;
  const int p0   = tile * 16;
  if (p0 >= npairs) return;                      // wave-uniform: EXEC stays full

  const int  arow = p0 + l15;                    // this lane gathers A row M=l15
  const bool rv   = arow < npairs;
  const long xrow = rv ? (long)gi[arow] : 0;
  const float* xp = X + xrow * (long)Cin;

  v8f acc = {};
  for (int k0 = 0, chunk = 0; k0 < Cin; k0 += 32, ++chunk) {
    v16h a;
    if (rv && (k0 + 32 <= Cin)) {                // fast path: 4x float4 gathers
      const float* pa = xp + k0 + (half << 3);
      float4 f0 = ((const float4*)pa)[0];
      float4 f1 = ((const float4*)pa)[1];
      float4 f2 = ((const float4*)(pa + 16))[0];
      float4 f3 = ((const float4*)(pa + 16))[1];
      a[0]  = (_Float16)f0.x; a[1]  = (_Float16)f0.y;
      a[2]  = (_Float16)f0.z; a[3]  = (_Float16)f0.w;
      a[4]  = (_Float16)f1.x; a[5]  = (_Float16)f1.y;
      a[6]  = (_Float16)f1.z; a[7]  = (_Float16)f1.w;
      a[8]  = (_Float16)f2.x; a[9]  = (_Float16)f2.y;
      a[10] = (_Float16)f2.z; a[11] = (_Float16)f2.w;
      a[12] = (_Float16)f3.x; a[13] = (_Float16)f3.y;
      a[14] = (_Float16)f3.z; a[15] = (_Float16)f3.w;
    } else {                                     // ragged tail / invalid row
#pragma unroll
      for (int j = 0; j < 16; ++j) {
        int v = j >> 1, p = j & 1;
        int kk = ((v >> 2) << 4) + (half << 3) + ((v & 3) << 1) + p;
        int c = k0 + kk;
        a[j] = (rv && c < Cin) ? (_Float16)xp[c] : (_Float16)0.0f;
      }
    }
    v16h b = *(const v16h*)&sB[(chunk << 9) + (lane << 4)];
    acc = __builtin_amdgcn_wmma_f32_16x16x32_f16(
        false, a, false, b, (short)0, acc, false, false);
  }
#pragma unroll
  for (int r = 0; r < 8; ++r) {                  // C/D: M = r + 8*half, N = l15
    int orow = p0 + (half << 3) + r;
    if (orow < npairs) {
      long d = (long)so[orow] * Cout + cout0 + l15;
      OUT[d] += acc[r];
    }
  }
}

// ---- deterministic per-channel mean/var (one block per channel) -----------
__global__ void __launch_bounds__(256) k_colstats(
    const float* __restrict__ X, int N, int C, float* __restrict__ stats)
{
  const int c = blockIdx.x;
  float s = 0.f, s2 = 0.f;
  for (int i = threadIdx.x; i < N; i += 256) {
    float v = X[(long)i * C + c];
    s += v; s2 += v * v;
  }
  __shared__ float sh[256], sq[256];
  sh[threadIdx.x] = s; sq[threadIdx.x] = s2;
  __syncthreads();
  for (int off = 128; off > 0; off >>= 1) {
    if (threadIdx.x < off) {
      sh[threadIdx.x] += sh[threadIdx.x + off];
      sq[threadIdx.x] += sq[threadIdx.x + off];
    }
    __syncthreads();
  }
  if (threadIdx.x == 0) {
    float mu  = sh[0] / (float)N;
    float var = sq[0] / (float)N - mu * mu;
    stats[2 * c] = mu;
    stats[2 * c + 1] = var;
  }
}

__global__ void __launch_bounds__(256) k_bnapply(
    const float* __restrict__ X, const float* __restrict__ stats,
    const float* __restrict__ g, const float* __restrict__ b,
    float* __restrict__ Y, long total, int C)
{
  long i = (long)blockIdx.x * 256 + threadIdx.x;
  if (i >= total) return;
  int c = (int)(i % C);
  float mu = stats[2 * c], var = stats[2 * c + 1];
  float y = (X[i] - mu) * (g[c] * rsqrtf(var + 1e-4f)) + b[c];
  Y[i] = y > 0.f ? y : 0.f;
}

__global__ void __launch_bounds__(256) k_add_inplace(
    float* __restrict__ A, const float* __restrict__ B, long n)
{
  long i = (long)blockIdx.x * 256 + threadIdx.x;
  if (i < n) A[i] += B[i];
}

__global__ void __launch_bounds__(256) k_add_out(
    const float* __restrict__ A, const float* __restrict__ B,
    float* __restrict__ O, long n)
{
  long i = (long)blockIdx.x * 256 + threadIdx.x;
  if (i < n) O[i] = A[i] + B[i];
}

__global__ void __launch_bounds__(256) k_concat2(
    const float* __restrict__ A, const float* __restrict__ B,
    float* __restrict__ O, long n, int c1, int c2)
{
  long i = (long)blockIdx.x * 256 + threadIdx.x;
  long tot = n * (long)(c1 + c2);
  if (i >= tot) return;
  long r = i / (c1 + c2);
  int  c = (int)(i % (c1 + c2));
  O[i] = (c < c1) ? A[r * c1 + c] : B[r * c2 + (c - c1)];
}

__global__ void __launch_bounds__(256) k_extract_feat(
    const float* __restrict__ PC, float* __restrict__ F, int n)
{
  int i = blockIdx.x * 256 + threadIdx.x;
  if (i < n) F[i] = PC[(long)i * 5 + 4];
}

__global__ void __launch_bounds__(256) k_iota(int* __restrict__ p, int n)
{
  int i = blockIdx.x * 256 + threadIdx.x;
  if (i < n) p[i] = i;
}

__global__ void __launch_bounds__(256) k_linear(
    const float* __restrict__ X, const float* __restrict__ W,
    const float* __restrict__ b, float* __restrict__ O, int N)
{
  int i = blockIdx.x * 256 + threadIdx.x;
  if (i >= N * 5) return;
  int n = i / 5, j = i % 5;
  float acc = b[j];
#pragma unroll
  for (int c = 0; c < 16; ++c) acc += X[(long)n * 16 + c] * W[c * 5 + j];
  O[i] = acc;
}

// ---------------------------------------------------------------------------
static inline void* bump(char* base, size_t& off, size_t bytes) {
  void* p = base + off;
  off += (bytes + 255) & ~(size_t)255;
  return p;
}
static inline unsigned gblk(long n) { return (unsigned)((n + 255) / 256); }

extern "C" void kernel_launch(void* const* d_in, const int* in_sizes, int n_in,
                              void* d_out, int out_size, void* d_ws, size_t ws_size,
                              hipStream_t stream)
{
  (void)n_in; (void)out_size; (void)ws_size;
  // ---- flatten map (insertion-order recursive; None skipped) --------------
  const float* pc   = (const float*)d_in[0];
  const float* w_in = (const float*)d_in[1];
  int idx = 2;
  const float* encB[NSTR][2][6];                       // g1,b1,w1,g2,b2,w2
  for (int i = 0; i < NSTR; ++i)
    for (int r = 0; r < 2; ++r)
      for (int j = 0; j < 6; ++j) encB[i][r][j] = (const float*)d_in[idx++];
  const float* dnP[NSTR - 1][3];                       // bn_g, bn_b, w
  for (int i = 0; i < NSTR - 1; ++i)
    for (int j = 0; j < 3; ++j) dnP[i][j] = (const float*)d_in[idx++];
  const float* decP[NSTR - 1][16];  // bg,bb,w, nin,b1g,b1b,w1,b2g,b2b,w2, (block2: g1,b1,w1,g2,b2,w2)
  for (int i = 0; i < NSTR - 1; ++i)
    for (int j = 0; j < 16; ++j) decP[i][j] = (const float*)d_in[idx++];
  const float* obn_g = (const float*)d_in[idx++];
  const float* obn_b = (const float*)d_in[idx++];
  const float* lin_w = (const float*)d_in[idx++];
  const float* lin_b = (const float*)d_in[idx++];
  const int *sub_ii[NSTR][27], *sub_oo[NSTR][27]; int sub_n[NSTR][27];
  for (int i = 0; i < NSTR; ++i)
    for (int k = 0; k < 27; ++k) {
      sub_ii[i][k] = (const int*)d_in[idx];
      sub_n[i][k]  = in_sizes[idx];
      sub_oo[i][k] = (const int*)d_in[idx + 1];
      idx += 2;
    }
  const int *dwn_fi[NSTR - 1][8], *dwn_ci[NSTR - 1][8]; int dwn_n[NSTR - 1][8];
  for (int i = 0; i < NSTR - 1; ++i)
    for (int k = 0; k < 8; ++k) {
      dwn_fi[i][k] = (const int*)d_in[idx];
      dwn_n[i][k]  = in_sizes[idx];
      dwn_ci[i][k] = (const int*)d_in[idx + 1];
      idx += 2;
    }
  int nsite[NSTR];
  for (int i = 0; i < NSTR; ++i) nsite[i] = in_sizes[idx + i];

  // ---- workspace ----------------------------------------------------------
  size_t mb = 0;
  for (int i = 0; i < NSTR; ++i) { size_t e = (size_t)nsite[i] * NP[i]; if (e > mb) mb = e; }
  for (int l = 0; l < NSTR - 1; ++l) { size_t e = (size_t)nsite[l] * 2 * NP[l]; if (e > mb) mb = e; }
  char* base = (char*)d_ws; size_t off = 0;
  float* B0 = (float*)bump(base, off, mb * 4);
  float* B1 = (float*)bump(base, off, mb * 4);
  float* B2 = (float*)bump(base, off, mb * 4);
  float* B3 = (float*)bump(base, off, mb * 4);
  float* fm[NSTR - 1];
  for (int l = 0; l < NSTR - 1; ++l) fm[l] = (float*)bump(base, off, (size_t)nsite[l] * NP[l] * 4);
  float* stats = (float*)bump(base, off, 512 * 4);
  int*   iot   = (int*)bump(base, off, (size_t)nsite[0] * 4);

  // ---- helpers ------------------------------------------------------------
  auto zero = [&](float* p, size_t elems) {
    hipMemsetAsync(p, 0, elems * sizeof(float), stream);
  };
  auto conv = [&](const float* Xp, const float* Wp, const int* gi, const int* so,
                  float* Op, int np_, int Ci, int Co) {
    if (np_ <= 0) return;
    int tiles = (np_ + 15) / 16;
    dim3 g((unsigned)((tiles + 7) / 8), (unsigned)(Co / 16));
    hipLaunchKernelGGL(k_spconv_wmma, g, dim3(256), 0, stream,
                       Xp, Wp, gi, so, Op, np_, Ci, Co);
  };
  auto bnrelu = [&](const float* Xp, const float* g, const float* bb,
                    float* Yp, int N, int C) {
    hipLaunchKernelGGL(k_colstats, dim3((unsigned)C), dim3(256), 0, stream, Xp, N, C, stats);
    long tot = (long)N * C;
    hipLaunchKernelGGL(k_bnapply, dim3(gblk(tot)), dim3(256), 0, stream,
                       Xp, stats, g, bb, Yp, tot, C);
  };
  auto subm = [&](const float* Xp, const float* W27, int lvl, float* Op,
                  int N, int Ci, int Co) {
    zero(Op, (size_t)N * Co);
    for (int k = 0; k < 27; ++k)
      conv(Xp, W27 + (size_t)k * Ci * Co, sub_ii[lvl][k], sub_oo[lvl][k],
           Op, sub_n[lvl][k], Ci, Co);
  };

  // ---- forward ------------------------------------------------------------
  hipLaunchKernelGGL(k_iota, dim3(gblk(nsite[0])), dim3(256), 0, stream, iot, nsite[0]);
  hipLaunchKernelGGL(k_extract_feat, dim3(gblk(nsite[0])), dim3(256), 0, stream, pc, B1, nsite[0]);
  subm(B1, w_in, 0, B0, nsite[0], 1, 16);              // x = subm(feats, w_in)

  for (int i = 0; i < NSTR; ++i) {                     // encoder
    int N = nsite[i], C = NP[i];
    for (int r = 0; r < 2; ++r) {                      // residual block, in-place on B0
      const float* const* P = encB[i][r];
      bnrelu(B0, P[0], P[1], B1, N, C);
      subm(B1, P[2], i, B2, N, C, C);
      bnrelu(B2, P[3], P[4], B1, N, C);
      subm(B1, P[5], i, B2, N, C, C);
      hipLaunchKernelGGL(k_add_inplace, dim3(gblk((long)N * C)), dim3(256), 0, stream,
                         B0, B2, (long)N * C);
    }
    if (i < NSTR - 1) {
      hipMemcpyAsync(fm[i], B0, (size_t)N * C * 4, hipMemcpyDeviceToDevice, stream);
      bnrelu(B0, dnP[i][0], dnP[i][1], B1, N, C);      // down-sample
      int N2 = nsite[i + 1], C2 = NP[i + 1];
      zero(B2, (size_t)N2 * C2);
      for (int k = 0; k < 8; ++k)
        conv(B1, dnP[i][2] + (size_t)k * C * C2, dwn_fi[i][k], dwn_ci[i][k],
             B2, dwn_n[i][k], C, C2);
      hipMemcpyAsync(B0, B2, (size_t)N2 * C2 * 4, hipMemcpyDeviceToDevice, stream);
    }
  }

  for (int i = 0; i < NSTR - 1; ++i) {                 // decoder
    int l  = NSTR - 2 - i;
    int Nc = nsite[l + 1], Cc = NP[l + 1];
    int Nf = nsite[l],     Cf = NP[l];
    const float* const* D = decP[i];
    bnrelu(B0, D[0], D[1], B1, Nc, Cc);                // bn_relu(x)
    zero(B2, (size_t)Nf * Cf);                         // up: gather coarse ci, write fine fi
    for (int k = 0; k < 8; ++k)
      conv(B1, D[2] + (size_t)k * Cc * Cf, dwn_ci[l][k], dwn_fi[l][k],
           B2, dwn_n[l][k], Cc, Cf);
    hipLaunchKernelGGL(k_concat2, dim3(gblk((long)Nf * 2 * Cf)), dim3(256), 0, stream,
                       fm[l], B2, B3, (long)Nf, Cf, Cf);
    hipMemcpyAsync(B0, B3, (size_t)Nf * 2 * Cf * 4, hipMemcpyDeviceToDevice, stream);
    {                                                  // block1: a=2Cf, b=Cf, with nin
      int a = 2 * Cf, b = Cf;
      zero(B3, (size_t)Nf * b);
      conv(B0, D[3], iot, iot, B3, Nf, a, b);          // shortcut = x @ nin (identity rulebook)
      bnrelu(B0, D[4], D[5], B1, Nf, a);
      subm(B1, D[6], l, B2, Nf, a, b);
      bnrelu(B2, D[7], D[8], B1, Nf, b);
      subm(B1, D[9], l, B2, Nf, b, b);
      hipLaunchKernelGGL(k_add_out, dim3(gblk((long)Nf * b)), dim3(256), 0, stream,
                         B3, B2, B0, (long)Nf * b);
    }
    {                                                  // block2: Cf -> Cf, in-place
      const float* const* P = D + 10;
      bnrelu(B0, P[0], P[1], B1, Nf, Cf);
      subm(B1, P[2], l, B2, Nf, Cf, Cf);
      bnrelu(B2, P[3], P[4], B1, Nf, Cf);
      subm(B1, P[5], l, B2, Nf, Cf, Cf);
      hipLaunchKernelGGL(k_add_inplace, dim3(gblk((long)Nf * Cf)), dim3(256), 0, stream,
                         B0, B2, (long)Nf * Cf);
    }
  }

  bnrelu(B0, obn_g, obn_b, B1, nsite[0], 16);          // head
  hipLaunchKernelGGL(k_linear, dim3(gblk((long)nsite[0] * 5)), dim3(256), 0, stream,
                     B1, lin_w, lin_b, (float*)d_out, nsite[0]);
}
